// Custom_Matmul_13228499272231
// MI455X (gfx1250) — 8-bit matrix operand order repaired and checked by reading, compiled, not run on hardware
//
#include <hip/hip_runtime.h>
#include <stdint.h>

typedef __attribute__((ext_vector_type(8))) int v8i;

#define H_   16
#define M_   2048
#define N_   2048
#define K_   128
#define BM   128
#define BN   128
#define LDSW 33

__global__ __launch_bounds__(256)
void int8_gemm_requant_kernel(const int*   __restrict__ Ag,
                              const int*   __restrict__ Bg,
                              const float* __restrict__ scaleA,
                              const float* __restrict__ scaleB,
                              const float* __restrict__ scaleO,
                              int*         __restrict__ Cg)
{
    __shared__ uint32_t lA[BM * LDSW];
    __shared__ uint32_t lB[BN * LDSW];

    const int h      = blockIdx.z;
    const int blockM = blockIdx.y * BM;
    const int blockN = blockIdx.x * BN;
    const int t      = threadIdx.x;

    {
        const size_t aBase = ((size_t)h * M_ + blockM) * K_;
        const size_t bBase = ((size_t)h * N_ + blockN) * K_;
#pragma unroll
        for (int it = 0; it < 16; ++it) {
            const int q  = t + it * 256;
            const int r  = q >> 5;
            const int kq = q & 31;
            const int4 wa = *(const int4*)(Ag + aBase + (size_t)r * K_ + kq * 4);
            const int4 wb = *(const int4*)(Bg + bBase + (size_t)r * K_ + kq * 4);
            const uint32_t pa = (uint32_t)(wa.x & 0xFF)        | ((uint32_t)(wa.y & 0xFF) << 8) |
                                ((uint32_t)(wa.z & 0xFF) << 16) | ((uint32_t)(wa.w & 0xFF) << 24);
            const uint32_t pb = (uint32_t)(wb.x & 0xFF)        | ((uint32_t)(wb.y & 0xFF) << 8) |
                                ((uint32_t)(wb.z & 0xFF) << 16) | ((uint32_t)(wb.w & 0xFF) << 24);
            lA[r * LDSW + kq] = pa;
            lB[r * LDSW + kq] = pb;
        }
    }
    __syncthreads();

    const int wave   = t >> 5;
    const int lane   = t & 31;
    const int laneLo = lane & 15;
    const int hi     = lane >> 4;
    const int mWave  = (wave >> 2) * 64;
    const int nWave  = (wave & 3) * 32;

    v8i acc[4][2] = {};

#pragma unroll
    for (int k0 = 0; k0 < K_; k0 += 64) {
        v8i af[4];
#pragma unroll
        for (int i = 0; i < 4; ++i) {
            const uint32_t* rowp = &lA[(mWave + 16 * i + laneLo) * LDSW];
#pragma unroll
            for (int v = 0; v < 8; ++v) {
                const int k = k0 + 16 * (v >> 1) + 8 * hi + 4 * (v & 1);
                af[i][v] = (int)rowp[k >> 2];
            }
        }
        v8i bf[2];
#pragma unroll
        for (int j = 0; j < 2; ++j) {
            const uint32_t* colp = &lB[(nWave + 16 * j + laneLo) * LDSW];
#pragma unroll
            for (int v = 0; v < 8; ++v) {
                const int k = k0 + 16 * (v >> 1) + 8 * hi + 4 * (v & 1);
                bf[j][v] = (int)colp[k >> 2];
            }
        }
#pragma unroll
        for (int i = 0; i < 4; ++i)
#pragma unroll
            for (int j = 0; j < 2; ++j)
                acc[i][j] = __builtin_amdgcn_wmma_i32_16x16x64_iu8(
                     false, af[i],  false, bf[j],
                    acc[i][j],  false,  false);
    }

    const float* sAh = scaleA + (size_t)h * M_;
    const float* sOh = scaleO + (size_t)h * M_;
    const float* sBh = scaleB + (size_t)h * N_;
    int*         Ch  = Cg     + (size_t)h * M_ * N_;

#pragma unroll
    for (int i = 0; i < 4; ++i) {
        const int mBase = blockM + mWave + 16 * i + 8 * hi;
#pragma unroll
        for (int r = 0; r < 8; ++r) {
            const int   m  = mBase + r;
            const float rs = sAh[m] / sOh[m];
#pragma unroll
            for (int j = 0; j < 2; ++j) {
                const int   n  = blockN + nWave + 16 * j + laneLo;
                const float sB = sBh[n];
                float f  = (float)acc[i][j][r] * (rs * sB);
                float rv = __builtin_rintf(f);
                rv = fminf(127.0f, fmaxf(-127.0f, rv));
                __builtin_nontemporal_store((int)rv, &Ch[(size_t)m * N_ + n]);
            }
        }
    }
}

__global__ void copy_scales_kernel(const float* __restrict__ s,
                                   float* __restrict__ o, int n)
{
    const int i = blockIdx.x * blockDim.x + threadIdx.x;
    if (i < n) o[i] = s[i];
}

extern "C" void kernel_launch(void* const* d_in, const int* in_sizes, int n_in,
                              void* d_out, int out_size, void* d_ws, size_t ws_size,
                              hipStream_t stream)
{
    const int*   A  = (const int*)d_in[0];
    const int*   B  = (const int*)d_in[1];
    const float* sA = (const float*)d_in[2];
    const float* sB = (const float*)d_in[3];
    const float* sO = (const float*)d_in[4];

    int*   outC = (int*)d_out;
    float* outS = (float*)(outC + (size_t)H_ * M_ * N_);

    dim3 grid(N_ / BN, M_ / BM, H_);
    int8_gemm_requant_kernel<<<grid, 256, 0, stream>>>(A, B, sA, sB, sO, outC);

    const int ns = H_ * M_;
    copy_scales_kernel<<<(ns + 255) / 256, 256, 0, stream>>>(sO, outS, ns);
}
